// Violin_Transformer_20675972563547
// MI455X (gfx1250) — matrix operand order repaired and checked by reading, not compiled, not run on hardware
//
#include <hip/hip_runtime.h>
#include <hip/hip_bf16.h>
#include <math.h>

// ---------------- model constants ----------------
#define PSZ     16
#define IMGD    224
#define CDIM    768
#define NHEAD   12
#define DEPTHL  12
#define SGRID   14
#define NPATCH  196
#define NTOK    197
#define HIDD    3072
#define NCLS    1000
#define BATCHN  32
#define NCURVE  6
#define HDIM    64              // head dim
#define ROWS    (BATCHN*NTOK)   // 6304
#define KPAD    224             // 197 padded to multiple of 32

typedef __bf16 bf16_t;
typedef __bf16          v16bf  __attribute__((ext_vector_type(16)));
typedef float           v8f    __attribute__((ext_vector_type(8)));
typedef unsigned short  u16x8  __attribute__((ext_vector_type(8)));
typedef unsigned short  u16x16 __attribute__((ext_vector_type(16)));

__device__ __forceinline__ bf16_t f2bf(float f) {
    unsigned u; __builtin_memcpy(&u, &f, 4);
    unsigned r = u + 0x7FFFu + ((u >> 16) & 1u);
    unsigned short h = (unsigned short)(r >> 16);
    bf16_t b; __builtin_memcpy(&b, &h, 2); return b;
}

__device__ __forceinline__ u16x16 frag_pair(const unsigned short* p0, const unsigned short* p1) {
    u16x8 lo = *(const u16x8*)p0;
    u16x8 hi = *(const u16x8*)p1;
    u16x16 r;
    #pragma unroll
    for (int j = 0; j < 8; ++j) { r[j] = lo[j]; r[j + 8] = hi[j]; }
    return r;
}

// ---------------- generic bf16 WMMA GEMM (32x32 macro-tile per wave) ----------------
// C[m,n] = sum_k A[m,k] * B[n,k]. A: row m at A + m*ldam, k contiguous.
// B: row n at B + n*ldbn, k contiguous. Out-of-range rows/cols are clamped on
// load (garbage lands only in store-masked D entries), keeping EXEC untouched.
// MODE: 0 = f32 out (+bias); 1 = f32 out (+bias+residual);
//       2 = bf16 out (+bias); 3 = bf16 out gelu(acc+bias)
template<int MODE>
__global__ void __launch_bounds__(256)
gemm_kernel(const bf16_t* __restrict__ A, long long ldam,
            const bf16_t* __restrict__ B, long long ldbn,
            const float* __restrict__ bias,
            float* __restrict__ outF, bf16_t* __restrict__ outB, long long ldo,
            const float* __restrict__ res, long long ldr,
            int M, int Nn, int K, int Hd,
            long long aB, long long aH, long long bB, long long bH,
            long long oB, long long oH, long long rB, long long rH)
{
    int z  = blockIdx.z;
    int bz = z / Hd, hz = z - bz * Hd;
    const bf16_t* Ab = A + (long long)bz * aB + (long long)hz * aH;
    const bf16_t* Bb = B + (long long)bz * bB + (long long)hz * bH;
    long long oOff = (long long)bz * oB + (long long)hz * oH;
    long long rOff = (long long)bz * rB + (long long)hz * rH;

    int Mt = (M + 31) >> 5, Nt = (Nn + 31) >> 5;
    int tile = blockIdx.x * (blockDim.x >> 5) + (threadIdx.x >> 5);
    if (tile >= Mt * Nt) return;              // wave-uniform
    int mt = tile / Nt, nt = tile - mt * Nt;
    int m0 = mt << 5, n0 = nt << 5;

    int lane = threadIdx.x & 31, g = lane >> 4, r = lane & 15;

    // clamped row/col pointers (garbage lands only in store-masked D entries)
    int ar0 = min(m0 + r,      M - 1);
    int ar1 = min(m0 + 16 + r, M - 1);
    const unsigned short* A0 = (const unsigned short*)(Ab + (long long)ar0 * ldam);
    const unsigned short* A1 = (const unsigned short*)(Ab + (long long)ar1 * ldam);
    int bn0 = min(n0 + r,      Nn - 1);
    int bn1 = min(n0 + 16 + r, Nn - 1);
    const unsigned short* B0 = (const unsigned short*)(Bb + (long long)bn0 * ldbn);
    const unsigned short* B1 = (const unsigned short*)(Bb + (long long)bn1 * ldbn);

    v8f acc00 = (v8f)0.f, acc01 = (v8f)0.f, acc10 = (v8f)0.f, acc11 = (v8f)0.f;

    for (int k0 = 0; k0 < K; k0 += 32) {
        u16x16 ra0 = frag_pair(A0 + k0 + g * 8, A0 + k0 + 16 + g * 8);
        u16x16 ra1 = frag_pair(A1 + k0 + g * 8, A1 + k0 + 16 + g * 8);
        __builtin_prefetch((const void*)(A0 + k0 + 32), 0, 1);
        __builtin_prefetch((const void*)(A1 + k0 + 32), 0, 1);
        u16x16 rb0 = frag_pair(B0 + k0 + g * 8, B0 + k0 + 16 + g * 8);
        u16x16 rb1 = frag_pair(B1 + k0 + g * 8, B1 + k0 + 16 + g * 8);
        v16bf a0 = __builtin_bit_cast(v16bf, ra0);
        v16bf a1 = __builtin_bit_cast(v16bf, ra1);
        v16bf b0 = __builtin_bit_cast(v16bf, rb0);
        v16bf b1 = __builtin_bit_cast(v16bf, rb1);
        acc00 = __builtin_amdgcn_wmma_f32_16x16x32_bf16(false, a0, false, b0, (short)0, acc00, false, false);
        acc01 = __builtin_amdgcn_wmma_f32_16x16x32_bf16(false, a0, false, b1, (short)0, acc01, false, false);
        acc10 = __builtin_amdgcn_wmma_f32_16x16x32_bf16(false, a1, false, b0, (short)0, acc10, false, false);
        acc11 = __builtin_amdgcn_wmma_f32_16x16x32_bf16(false, a1, false, b1, (short)0, acc11, false, false);
    }

    // ---- epilogue: store with true-index guards ----
    auto store_tile = [&](v8f acc, int mbase, int nIdx) {
        if (nIdx >= Nn) return;
        float bv = bias ? bias[nIdx] : 0.f;
        #pragma unroll
        for (int i = 0; i < 8; ++i) {
            int m = mbase + g * 8 + i;
            if (m >= M) continue;
            float v = acc[i] + bv;
            long long o = oOff + (long long)m * ldo + nIdx;
            if (MODE == 0) {
                outF[o] = v;
            } else if (MODE == 1) {
                outF[o] = v + res[rOff + (long long)m * ldr + nIdx];
            } else if (MODE == 2) {
                outB[o] = f2bf(v);
            } else {
                float ge = 0.5f * v * (1.f + erff(v * 0.70710678118654752f));
                outB[o] = f2bf(ge);
            }
        }
    };
    store_tile(acc00, m0,      n0 + r);
    store_tile(acc01, m0,      n0 + 16 + r);
    store_tile(acc10, m0 + 16, n0 + r);
    store_tile(acc11, m0 + 16, n0 + 16 + r);
}

// ---------------- LayerNorm (f32 in -> bf16 out) ----------------
__global__ void __launch_bounds__(256)
ln_kernel(const float* __restrict__ x, const float* __restrict__ w,
          const float* __restrict__ b, bf16_t* __restrict__ out)
{
    __shared__ float red[256];
    int row = blockIdx.x;
    const float* xr = x + (long long)row * CDIM;
    int t = threadIdx.x;
    float s = 0.f;
    for (int c = t; c < CDIM; c += 256) s += xr[c];
    red[t] = s; __syncthreads();
    for (int k = 128; k > 0; k >>= 1) { if (t < k) red[t] += red[t + k]; __syncthreads(); }
    float mean = red[0] * (1.f / CDIM); __syncthreads();
    float v = 0.f;
    for (int c = t; c < CDIM; c += 256) { float d = xr[c] - mean; v += d * d; }
    red[t] = v; __syncthreads();
    for (int k = 128; k > 0; k >>= 1) { if (t < k) red[t] += red[t + k]; __syncthreads(); }
    float rstd = rsqrtf(red[0] * (1.f / CDIM) + 1e-5f);
    for (int c = t; c < CDIM; c += 256)
        out[(long long)row * CDIM + c] = f2bf((xr[c] - mean) * rstd * w[c] + b[c]);
}

// ---------------- curve distance matrix ----------------
__global__ void dmat_kernel(const int* __restrict__ ci, float* __restrict__ D)
{
    int idx = blockIdx.x * blockDim.x + threadIdx.x;
    int total = NCURVE * NPATCH * NPATCH;
    if (idx >= total) return;
    int q = idx % NPATCH;
    int p = (idx / NPATCH) % NPATCH;
    int c = idx / (NPATCH * NPATCH);
    D[idx] = fabsf((float)(ci[c * NPATCH + p] - ci[c * NPATCH + q]));
}

// ---------------- per-layer attention mask ----------------
__global__ void mask_kernel(const float* __restrict__ ai_l, const float* __restrict__ D,
                            float* __restrict__ Mm)
{
    long long idx = (long long)blockIdx.x * blockDim.x + threadIdx.x;
    long long total = (long long)NHEAD * NTOK * NTOK;
    if (idx >= total) return;
    int j = (int)(idx % NTOK);
    int i = (int)((idx / NTOK) % NTOK);
    int h = (int)(idx / ((long long)NTOK * NTOK));
    float outv = 1.f;
    if (i > 0 && j > 0) {
        float s = 0.f;
        #pragma unroll
        for (int c = 0; c < NCURVE; ++c) {
            float a   = ai_l[c * NHEAD + h];
            float sig = 1.f / (1.f + __expf(-a));
            float d   = D[((long long)c * NPATCH + (i - 1)) * NPATCH + (j - 1)];
            s += __expf(d * __logf(sig));
        }
        outv = s * (1.f / NCURVE);
    }
    Mm[idx] = outv;
}

// ---------------- masked softmax (f32 scores -> bf16 P, K-padded) ----------------
__global__ void __launch_bounds__(256)
masked_softmax_kernel(const float* __restrict__ S, const float* __restrict__ Mm,
                      bf16_t* __restrict__ P, float scale)
{
    __shared__ float red[256];
    int row = blockIdx.x;                        // (b*NHEAD + h)*NTOK + i
    int i   = row % NTOK;
    int h   = (row / NTOK) % NHEAD;
    const float* sr = S + (long long)row * NTOK;
    const float* mr = Mm + ((long long)h * NTOK + i) * NTOK;
    bf16_t* pr = P + (long long)row * KPAD;
    int t = threadIdx.x;
    float v = (t < NTOK) ? sr[t] * scale * mr[t] : -3.0e38f;
    red[t] = v; __syncthreads();
    for (int k = 128; k > 0; k >>= 1) { if (t < k) red[t] = fmaxf(red[t], red[t + k]); __syncthreads(); }
    float mx = red[0]; __syncthreads();
    float e = (t < NTOK) ? __expf(v - mx) : 0.f;
    red[t] = e; __syncthreads();
    for (int k = 128; k > 0; k >>= 1) { if (t < k) red[t] += red[t + k]; __syncthreads(); }
    float inv = 1.f / red[0];
    if (t < KPAD) pr[t] = (t < NTOK) ? f2bf(e * inv) : f2bf(0.f);
}

// ---------------- V repack: vt[b,h,n,k] = V[b,k,h,n], K padded with zeros ----------------
__global__ void vtrans_kernel(const bf16_t* __restrict__ qkvb, bf16_t* __restrict__ vt)
{
    long long idx = (long long)blockIdx.x * blockDim.x + threadIdx.x;
    long long total = (long long)BATCHN * NHEAD * HDIM * KPAD;
    if (idx >= total) return;
    int k  = (int)(idx % KPAD);
    long long t1 = idx / KPAD;
    int n  = (int)(t1 % HDIM);
    long long t2 = t1 / HDIM;
    int h  = (int)(t2 % NHEAD);
    int b  = (int)(t2 / NHEAD);
    bf16_t v = f2bf(0.f);
    if (k < NTOK)
        v = qkvb[((long long)b * NTOK + k) * (3 * CDIM) + 2 * CDIM + h * HDIM + n];
    vt[idx] = v;
}

// ---------------- im2col for patch embedding ----------------
__global__ void im2col_kernel(const float* __restrict__ img, bf16_t* __restrict__ out)
{
    long long idx = (long long)blockIdx.x * blockDim.x + threadIdx.x;
    long long total = (long long)BATCHN * NPATCH * CDIM;
    if (idx >= total) return;
    int col = (int)(idx % CDIM);
    long long row = idx / CDIM;
    int b = (int)(row / NPATCH);
    int p = (int)(row % NPATCH);
    int py = p / SGRID, px = p % SGRID;
    int ci = col >> 8;         // / 256
    int rem = col & 255;
    int ky = rem >> 4, kx = rem & 15;
    long long src = (((long long)b * 3 + ci) * IMGD + (py * PSZ + ky)) * IMGD + (px * PSZ + kx);
    out[idx] = f2bf(img[src]);
}

// ---------------- assemble x = [cls ; patches] + pos ----------------
__global__ void assemble_kernel(const float* __restrict__ tmp, const float* __restrict__ cls,
                                const float* __restrict__ pos, float* __restrict__ x)
{
    long long idx = (long long)blockIdx.x * blockDim.x + threadIdx.x;
    long long total = (long long)ROWS * CDIM;
    if (idx >= total) return;
    int c = (int)(idx % CDIM);
    long long row = idx / CDIM;
    int n = (int)(row % NTOK);
    int b = (int)(row / NTOK);
    float v;
    if (n == 0) v = cls[c] + pos[c];
    else        v = tmp[((long long)b * NPATCH + (n - 1)) * CDIM + c] + pos[(long long)n * CDIM + c];
    x[idx] = v;
}

// ---------------- f32 -> bf16 weight convert (optional transpose) ----------------
__global__ void convt_kernel(const float* __restrict__ in, bf16_t* __restrict__ out,
                             int Nn, int K, int transpose)
{
    long long idx = (long long)blockIdx.x * blockDim.x + threadIdx.x;
    long long total = (long long)Nn * K;
    if (idx >= total) return;
    int n = (int)(idx / K);
    int k = (int)(idx % K);
    float v = transpose ? in[(long long)k * Nn + n] : in[idx];
    out[idx] = f2bf(v);
}

// ---------------- GEMM dispatch ----------------
static void launch_gemm(hipStream_t stream,
                        const bf16_t* A, long long ldam,
                        const bf16_t* B, long long ldbn,
                        const float* bias, float* oF, bf16_t* oBf, long long ldo,
                        const float* res, long long ldr,
                        int M, int Nn, int K, int mode,
                        int nz, int Hd,
                        long long aBs, long long aHs, long long bBs, long long bHs,
                        long long oBs, long long oHs, long long rBs, long long rHs)
{
    int Mt = (M + 31) / 32, Nt = (Nn + 31) / 32;
    int blocks = (Mt * Nt + 7) / 8;
    dim3 grid(blocks, 1, nz);
#define GEMM_ARGS A, ldam, B, ldbn, bias, oF, oBf, ldo, res, ldr, \
                  M, Nn, K, Hd, aBs, aHs, bBs, bHs, oBs, oHs, rBs, rHs
    switch (mode) {
    case 0:  gemm_kernel<0><<<grid, 256, 0, stream>>>(GEMM_ARGS); break;
    case 1:  gemm_kernel<1><<<grid, 256, 0, stream>>>(GEMM_ARGS); break;
    case 2:  gemm_kernel<2><<<grid, 256, 0, stream>>>(GEMM_ARGS); break;
    default: gemm_kernel<3><<<grid, 256, 0, stream>>>(GEMM_ARGS); break;
    }
#undef GEMM_ARGS
}

// ---------------- host orchestration ----------------
extern "C" void kernel_launch(void* const* d_in, const int* in_sizes, int n_in,
                              void* d_out, int out_size, void* d_ws, size_t ws_size,
                              hipStream_t stream)
{
    (void)in_sizes; (void)n_in; (void)out_size; (void)ws_size;
    const float* images  = (const float*)d_in[0];
    const float* patch_w = (const float*)d_in[1];
    const float* patch_b = (const float*)d_in[2];
    const float* cls_tok = (const float*)d_in[3];
    const float* pos_emb = (const float*)d_in[4];
    const float* ln1_w   = (const float*)d_in[5];
    const float* ln1_b   = (const float*)d_in[6];
    const float* qkv_w   = (const float*)d_in[7];
    const float* proj_w  = (const float*)d_in[8];
    const float* proj_b  = (const float*)d_in[9];
    const float* ai      = (const float*)d_in[10];
    const float* ln2_w   = (const float*)d_in[11];
    const float* ln2_b   = (const float*)d_in[12];
    const float* fc1_w   = (const float*)d_in[13];
    const float* fc1_b   = (const float*)d_in[14];
    const float* fc2_w   = (const float*)d_in[15];
    const float* fc2_b   = (const float*)d_in[16];
    const float* norm_w  = (const float*)d_in[17];
    const float* norm_b  = (const float*)d_in[18];
    const float* head_w  = (const float*)d_in[19];
    const float* head_b  = (const float*)d_in[20];
    const int*   curvei  = (const int*)d_in[21];
    float* out = (float*)d_out;

    char* ws = (char*)d_ws;
    size_t off = 0;
    auto alloc = [&](size_t bytes) -> char* {
        char* p = ws + off; off += (bytes + 255) & ~(size_t)255; return p;
    };
    float*  x      = (float*) alloc((size_t)ROWS * CDIM * 4);
    bf16_t* hbuf   = (bf16_t*)alloc((size_t)ROWS * CDIM * 2);
    bf16_t* qkvb   = (bf16_t*)alloc((size_t)ROWS * 3 * CDIM * 2);
    float*  scores = (float*) alloc((size_t)BATCHN * NHEAD * NTOK * NTOK * 4);
    float*  Mm     = (float*) alloc((size_t)NHEAD * NTOK * NTOK * 4);
    bf16_t* Pbuf   = (bf16_t*)alloc((size_t)BATCHN * NHEAD * NTOK * KPAD * 2);
    bf16_t* vt     = (bf16_t*)alloc((size_t)BATCHN * NHEAD * HDIM * KPAD * 2);
    bf16_t* obuf   = (bf16_t*)alloc((size_t)ROWS * CDIM * 2);
    bf16_t* hid    = (bf16_t*)alloc((size_t)ROWS * HIDD * 2);
    bf16_t* wT     = (bf16_t*)alloc((size_t)HIDD * CDIM * 2);
    bf16_t* acol   = (bf16_t*)alloc((size_t)BATCHN * NPATCH * CDIM * 2);
    float*  Dm     = (float*) alloc((size_t)NCURVE * NPATCH * NPATCH * 4);

    // ---- prologue ----
    dmat_kernel<<<(NCURVE * NPATCH * NPATCH + 255) / 256, 256, 0, stream>>>(curvei, Dm);
    {
        long long t = (long long)BATCHN * NPATCH * CDIM;
        im2col_kernel<<<(unsigned)((t + 255) / 256), 256, 0, stream>>>(images, acol);
    }
    convt_kernel<<<(CDIM * CDIM + 255) / 256, 256, 0, stream>>>(patch_w, wT, CDIM, CDIM, 0);
    // patch GEMM -> tmp (reuse scores buffer)
    launch_gemm(stream, acol, CDIM, wT, CDIM, patch_b, scores, nullptr, CDIM,
                nullptr, 0, BATCHN * NPATCH, CDIM, CDIM, 0, 1, 1,
                0, 0, 0, 0, 0, 0, 0, 0);
    {
        long long t = (long long)ROWS * CDIM;
        assemble_kernel<<<(unsigned)((t + 255) / 256), 256, 0, stream>>>(scores, cls_tok, pos_emb, x);
    }

    const float scale = 0.125f;   // 64^-0.5
    for (int l = 0; l < DEPTHL; ++l) {
        // LN1
        ln_kernel<<<ROWS, 256, 0, stream>>>(x, ln1_w + l * CDIM, ln1_b + l * CDIM, hbuf);
        // qkv
        convt_kernel<<<(3 * CDIM * CDIM + 255) / 256, 256, 0, stream>>>(
            qkv_w + (size_t)l * CDIM * 3 * CDIM, wT, 3 * CDIM, CDIM, 1);
        launch_gemm(stream, hbuf, CDIM, wT, CDIM, nullptr, nullptr, qkvb, 3 * CDIM,
                    nullptr, 0, ROWS, 3 * CDIM, CDIM, 2, 1, 1,
                    0, 0, 0, 0, 0, 0, 0, 0);
        // attention mask for this layer
        mask_kernel<<<(NHEAD * NTOK * NTOK + 255) / 256, 256, 0, stream>>>(
            ai + l * NCURVE * NHEAD, Dm, Mm);
        // scores = Q . K^T per (b,h)
        launch_gemm(stream, qkvb, 3 * CDIM, qkvb + CDIM, 3 * CDIM, nullptr, scores, nullptr, NTOK,
                    nullptr, 0, NTOK, NTOK, HDIM, 0, BATCHN * NHEAD, NHEAD,
                    (long long)NTOK * 3 * CDIM, HDIM,
                    (long long)NTOK * 3 * CDIM, HDIM,
                    (long long)NHEAD * NTOK * NTOK, (long long)NTOK * NTOK, 0, 0);
        // masked softmax -> P (bf16, K-padded)
        masked_softmax_kernel<<<BATCHN * NHEAD * NTOK, 256, 0, stream>>>(scores, Mm, Pbuf, scale);
        // repack V -> vt[b,h,n,k] (bf16, padded) so P.V runs the vectorized path
        {
            long long t = (long long)BATCHN * NHEAD * HDIM * KPAD;
            vtrans_kernel<<<(unsigned)((t + 255) / 256), 256, 0, stream>>>(qkvb, vt);
        }
        // O = P . V per (b,h)
        launch_gemm(stream, Pbuf, KPAD, vt, KPAD, nullptr, nullptr, obuf, CDIM,
                    nullptr, 0, NTOK, HDIM, KPAD, 2, BATCHN * NHEAD, NHEAD,
                    (long long)NHEAD * NTOK * KPAD, (long long)NTOK * KPAD,
                    (long long)NHEAD * HDIM * KPAD, (long long)HDIM * KPAD,
                    (long long)NTOK * CDIM, HDIM, 0, 0);
        // proj + residual
        convt_kernel<<<(CDIM * CDIM + 255) / 256, 256, 0, stream>>>(
            proj_w + (size_t)l * CDIM * CDIM, wT, CDIM, CDIM, 1);
        launch_gemm(stream, obuf, CDIM, wT, CDIM, proj_b + l * CDIM, x, nullptr, CDIM,
                    x, CDIM, ROWS, CDIM, CDIM, 1, 1, 1,
                    0, 0, 0, 0, 0, 0, 0, 0);
        // LN2
        ln_kernel<<<ROWS, 256, 0, stream>>>(x, ln2_w + l * CDIM, ln2_b + l * CDIM, hbuf);
        // fc1 + gelu
        convt_kernel<<<(CDIM * HIDD + 255) / 256, 256, 0, stream>>>(
            fc1_w + (size_t)l * CDIM * HIDD, wT, HIDD, CDIM, 1);
        launch_gemm(stream, hbuf, CDIM, wT, CDIM, fc1_b + l * HIDD, nullptr, hid, HIDD,
                    nullptr, 0, ROWS, HIDD, CDIM, 3, 1, 1,
                    0, 0, 0, 0, 0, 0, 0, 0);
        // fc2 + residual
        convt_kernel<<<(CDIM * HIDD + 255) / 256, 256, 0, stream>>>(
            fc2_w + (size_t)l * HIDD * CDIM, wT, CDIM, HIDD, 1);
        launch_gemm(stream, hid, HIDD, wT, HIDD, fc2_b + l * CDIM, x, nullptr, CDIM,
                    x, CDIM, ROWS, CDIM, HIDD, 1, 1, 1,
                    0, 0, 0, 0, 0, 0, 0, 0);
    }

    // ---- epilogue: final LN + head on CLS rows ----
    ln_kernel<<<ROWS, 256, 0, stream>>>(x, norm_w, norm_b, hbuf);
    convt_kernel<<<(CDIM * NCLS + 255) / 256, 256, 0, stream>>>(head_w, wT, NCLS, CDIM, 1);
    launch_gemm(stream, hbuf, (long long)NTOK * CDIM, wT, CDIM, head_b, out, nullptr, NCLS,
                nullptr, 0, BATCHN, NCLS, CDIM, 0, 1, 1,
                0, 0, 0, 0, 0, 0, 0, 0);
}
